// DualVQVAE_28321014350522
// MI455X (gfx1250) — matrix operand order repaired and checked by reading, not compiled, not run on hardware
//
#include <hip/hip_runtime.h>

// ---------------------------------------------------------------------------
// Dual VQ-VAE forward for MI455X (gfx1250, wave32, WMMA).
// All heavy math runs through v_wmma_f32_16x16x32_bf16 (bf16 in, f32 acc).
// GEMM wave tile widened to 32Mx64N: 8 WMMAs per 32-k step, B fragments
// reused across two A fragments (~21 FLOP/byte from L2).
// ---------------------------------------------------------------------------

typedef __attribute__((ext_vector_type(16))) __bf16  bf16x16;
typedef __attribute__((ext_vector_type(8)))  float   f32x8;
typedef __attribute__((ext_vector_type(4)))  unsigned int u32x4;

#define DEVI static __device__ __forceinline__

DEVI unsigned short f2bf(float f) {            // RNE f32 -> bf16
  unsigned u = __float_as_uint(f);
  u += 0x7fffu + ((u >> 16) & 1u);
  return (unsigned short)(u >> 16);
}

// ---------------------------------------------------------------------------
// small helper kernels
// ---------------------------------------------------------------------------
__global__ void k_zero(float* p, int n) {
  int i = blockIdx.x * blockDim.x + threadIdx.x;
  if (i < n) p[i] = 0.f;
}

__global__ void k_cvt(const float* __restrict__ s, unsigned short* __restrict__ d, long long n) {
  long long i = (long long)blockIdx.x * blockDim.x + threadIdx.x;
  if (i < n) d[i] = f2bf(s[i]);
}

// ConvTranspose weights: src [I][O][K] -> dst bf16 [O][I*K] flipped along K
__global__ void k_wT(const float* __restrict__ w, unsigned short* __restrict__ out,
                     int I, int O, int Kk) {
  int i = blockIdx.x * blockDim.x + threadIdx.x;
  int total = I * O * Kk;
  if (i >= total) return;
  int kk = i % Kk; int rest = i / Kk; int ii = rest % I; int o = rest / I;
  out[(size_t)o * I * Kk + ii * Kk + kk] =
      f2bf(w[((size_t)ii * O + o) * Kk + (Kk - 1 - kk)]);
}

// BN fold: scale = g*rsqrt(v+eps); shift = b - m*scale + conv_bias*scale
__global__ void k_fold(const float* g, const float* bb, const float* mm, const float* vv,
                       const float* cb, float* scale, float* shift, int C) {
  int i = blockIdx.x * blockDim.x + threadIdx.x;
  if (i >= C) return;
  float s = g[i] * rsqrtf(vv[i] + 1e-5f);
  scale[i] = s;
  shift[i] = bb[i] - mm[i] * s + (cb ? cb[i] * s : 0.f);
}

// ---------------------------------------------------------------------------
// im2col (normal + lhs-dilated for ConvTranspose).  x is CN f32 [C][B*Lin].
// out bf16 [B*Lout][C*K] row-major (GEMM Bt operand).
// ---------------------------------------------------------------------------
__global__ void k_im2col(const float* __restrict__ x, unsigned short* __restrict__ out,
                         int C, int Kk, int stride, int pad, int Lin, int Lout, int Bn) {
  long long i = (long long)blockIdx.x * blockDim.x + threadIdx.x;
  int Kt = C * Kk;
  long long total = (long long)Bn * Lout * Kt;
  if (i >= total) return;
  int rr = (int)(i % Kt); long long n = i / Kt;
  int c = rr / Kk, kk = rr % Kk;
  int b = (int)(n / Lout), l = (int)(n % Lout);
  int t = l * stride + kk - pad;
  float v = 0.f;
  if (t >= 0 && t < Lin) v = x[(size_t)c * Bn * Lin + (size_t)b * Lin + t];
  out[(size_t)n * Kt + rr] = f2bf(v);
}

__global__ void k_im2col_dil(const float* __restrict__ x, unsigned short* __restrict__ out,
                             int C, int Kk, int padL, int Lin, int Lout, int Bn) {
  long long i = (long long)blockIdx.x * blockDim.x + threadIdx.x;
  int Kt = C * Kk;
  long long total = (long long)Bn * Lout * Kt;
  if (i >= total) return;
  int rr = (int)(i % Kt); long long n = i / Kt;
  int c = rr / Kk, kk = rr % Kk;
  int b = (int)(n / Lout), l = (int)(n % Lout);
  int d = l + kk - padL;          // coordinate in lhs-dilated (stride-2) input
  float v = 0.f;
  if (d >= 0 && d < 2 * Lin - 1 && !(d & 1))
    v = x[(size_t)c * Bn * Lin + (size_t)b * Lin + (d >> 1)];
  out[(size_t)n * Kt + rr] = f2bf(v);
}

// ---------------------------------------------------------------------------
// WMMA GEMM:  C[M,N] = A[M,K] * Bt[N,K]^T   (both bf16 row-major, K%32==0,
// M%64==0, N%256==0).  Wave -> 32Mx64N tile (8 WMMAs / 32-k step, B reuse x2);
// block(256) = 2x4 waves -> 64Mx256N tile.
// Fused epilogue: (scale,shift) or bias, residual add, LeakyReLU, dual
// f32/bf16 stores in CN (m*N+n) or NC (n*ld+m) layout.
// ---------------------------------------------------------------------------
__global__ __launch_bounds__(256)
void k_gemm(const unsigned short* __restrict__ A, const unsigned short* __restrict__ Bt,
            int M, int N, int K,
            const float* __restrict__ scale, const float* __restrict__ shift,
            const float* __restrict__ res, int res_nc,
            float* __restrict__ outf, int outf_nc,
            unsigned short* __restrict__ outb, int outb_nc, int outb_ld,
            int relu) {
  const int lane = threadIdx.x & 31;
  const int wv   = threadIdx.x >> 5;
  const int lo = lane & 15, hi = lane >> 4;
  const int mt = blockIdx.y * 4 + (wv >> 2) * 2;        // two consecutive m-tiles
  const int nb = (blockIdx.x * 4 + (wv & 3)) * 4;       // four n-tiles

  const unsigned short* arow0 = A + (size_t)(mt * 16 + lo) * K;
  const unsigned short* arow1 = arow0 + (size_t)16 * K;
  const unsigned short* bp[4];
#pragma unroll
  for (int t = 0; t < 4; t++)
    bp[t] = Bt + (size_t)((nb + t) * 16 + lo) * K + hi * 16;

  f32x8 acc[8] = {};
  for (int k = 0; k < K; k += 32) {
    union { u32x4 u[2]; bf16x16 v; } a0, a1;
    a0.u[0] = *(const u32x4*)(arow0 + k + hi * 16);
    a0.u[1] = *(const u32x4*)(arow0 + k + hi * 16 + 8);
    a1.u[0] = *(const u32x4*)(arow1 + k + hi * 16);
    a1.u[1] = *(const u32x4*)(arow1 + k + hi * 16 + 8);
#pragma unroll
    for (int t = 0; t < 4; t++) {
      bf16x16 bv = *(const bf16x16*)(bp[t] + k);
      acc[t]     = __builtin_amdgcn_wmma_f32_16x16x32_bf16(
          false, a0.v, false, bv, (short)0, acc[t], false, false);
      acc[4 + t] = __builtin_amdgcn_wmma_f32_16x16x32_bf16(
          false, a1.v, false, bv, (short)0, acc[4 + t], false, false);
    }
  }
#pragma unroll
  for (int p = 0; p < 2; p++) {
#pragma unroll
    for (int t = 0; t < 4; t++) {
      size_t n0 = (size_t)(nb + t) * 16 + lo;
#pragma unroll
      for (int r = 0; r < 8; r++) {
        int m = (mt + p) * 16 + r + 8 * hi;
        float y = acc[p * 4 + t][r];
        if (scale)      y = y * scale[m] + shift[m];
        else if (shift) y += shift[m];
        if (res)  y += res[res_nc ? n0 * M + m : (size_t)m * N + n0];
        if (relu) y = (y < 0.f) ? 0.1f * y : y;
        if (outf) outf[outf_nc ? n0 * M + m : (size_t)m * N + n0] = y;
        if (outb) outb[outb_nc ? n0 * outb_ld + m : (size_t)m * N + n0] = f2bf(y);
      }
    }
  }
}

// ---------------------------------------------------------------------------
// encoder conv0 (Cin=1,K=5,s=2,p=2) direct + BN + LeakyReLU. out CN [256][B*4096]
// ---------------------------------------------------------------------------
__global__ void k_conv0(const float* __restrict__ x, const float* __restrict__ w,
                        const float* __restrict__ scale, const float* __restrict__ shift,
                        float* __restrict__ out, int Bn) {
  const int Nn = Bn * 4096;
  long long i = (long long)blockIdx.x * blockDim.x + threadIdx.x;
  if (i >= (long long)256 * Nn) return;
  int n = (int)(i % Nn); int m = (int)(i / Nn);
  int b = n >> 12, l = n & 4095;
  float a = 0.f;
#pragma unroll
  for (int kk = 0; kk < 5; kk++) {
    int t = l * 2 + kk - 2;
    if (t >= 0 && t < 8192) a += w[m * 5 + kk] * x[(size_t)b * 8192 + t];
  }
  float y = a * scale[m] + shift[m];
  out[(size_t)m * Nn + n] = (y < 0.f) ? 0.1f * y : y;
}

// ---------------------------------------------------------------------------
// decoder final ConvTranspose (I=256,O=1,K=5,s=2,p=2,op=1) direct. out [B*8192]
// ---------------------------------------------------------------------------
__global__ void k_d2(const float* __restrict__ x, const float* __restrict__ w,
                     const float* __restrict__ bias, float* __restrict__ out, int Bn) {
  const int Lin = 4096;
  int i = blockIdx.x * blockDim.x + threadIdx.x;
  if (i >= Bn * 8192) return;
  int b = i >> 13, l = i & 8191;
  float a = bias[0];
  for (int c = 0; c < 256; c++) {
    const float* xc = x + (size_t)c * Bn * Lin + (size_t)b * Lin;
#pragma unroll
    for (int kk = 0; kk < 5; kk++) {
      int d = l + kk - 2;                       // padL = K-1-p = 2
      if (d >= 0 && d < 2 * Lin - 1 && !(d & 1))
        a += w[c * 5 + (4 - kk)] * xc[d >> 1];  // flipped kernel
    }
  }
  out[i] = a;
}

// ---------------------------------------------------------------------------
// VQ: codebook norms, argmin over (|e|^2 - 2 dot), gather + loss reduction
// ---------------------------------------------------------------------------
__global__ void k_enorm(const float* __restrict__ e, float* __restrict__ out) {
  int m = blockIdx.x * blockDim.x + threadIdx.x;
  if (m >= 1024) return;
  float s = 0.f;
  for (int c = 0; c < 512; c++) { float v = e[(size_t)m * 512 + c]; s += v * v; }
  out[m] = s;
}

__global__ void k_argmin(const float* __restrict__ dot, const float* __restrict__ enorm,
                         int* __restrict__ idx, int N) {
  int n = blockIdx.x * blockDim.x + threadIdx.x;
  if (n >= N) return;
  float best = 3.0e38f; int bi = 0;
  for (int m = 0; m < 1024; m++) {
    float d = enorm[m] - 2.f * dot[(size_t)m * N + n];
    if (d < best) { best = d; bi = m; }
  }
  idx[n] = bi;
}

__global__ __launch_bounds__(256)
void k_gather(const int* __restrict__ idx, const float* __restrict__ emb,
              const float* __restrict__ z, unsigned short* __restrict__ qb,
              float* __restrict__ loss, float coef, int total) {
  int i = blockIdx.x * blockDim.x + threadIdx.x;
  float d2 = 0.f;
  if (i < total) {
    int n = i >> 9, c = i & 511;
    float q = emb[(size_t)idx[n] * 512 + c];
    qb[i] = f2bf(q);
    float df = q - z[i];
    d2 = df * df;
  }
  for (int o = 16; o; o >>= 1) d2 += __shfl_down(d2, o, 32);
  if ((threadIdx.x & 31) == 0) atomicAdd(loss, d2 * coef);
}

// V transpose: qkv[n][1536] (V at +1024) -> Vt[b][h][d][S]
__global__ void k_vt(const unsigned short* __restrict__ qkv, unsigned short* __restrict__ vt) {
  long long i = (long long)blockIdx.x * blockDim.x + threadIdx.x;
  if (i >= (long long)16 * 8 * 64 * 1024) return;
  int s = (int)(i & 1023);
  int d = (int)((i >> 10) & 63);
  int h = (int)((i >> 16) & 7);
  int b = (int)(i >> 19);
  vt[i] = qkv[(size_t)(b * 1024 + s) * 1536 + 1024 + h * 64 + d];
}

// ---------------------------------------------------------------------------
// Fused flash attention: one wave per (b, h, 16-query tile), k streamed in
// chunks of 32 with online softmax.  Scores computed K-major so the exp'd
// probabilities repack directly into the P.V A-fragment (no LDS).
// ---------------------------------------------------------------------------
__global__ __launch_bounds__(256)
void k_attn(const unsigned short* __restrict__ Qt, const unsigned short* __restrict__ Kt,
            const unsigned short* __restrict__ Vt, unsigned short* __restrict__ O) {
  const int S = 1024;
  int wave = blockIdx.x * 8 + (threadIdx.x >> 5);
  int lane = threadIdx.x & 31;
  int lo = lane & 15, hi = lane >> 4;
  int qt = wave & 63, h = (wave >> 6) & 7, b = wave >> 9;

  const unsigned short* qb = Qt + (size_t)(b * S + qt * 16 + lo) * 1536 + h * 64 + hi * 16;
  bf16x16 qf0 = *(const bf16x16*)(qb);        // d 0..31
  bf16x16 qf1 = *(const bf16x16*)(qb + 32);   // d 32..63

  float rM = -3.0e38f, rS = 0.f;
  f32x8 o0 = {}, o1 = {}, o2 = {}, o3 = {};

  for (int kc = 0; kc < S; kc += 32) {
    f32x8 s0 = {}, s1 = {};
    const unsigned short* kb0 = Kt + (size_t)(b * S + kc + (lo & 7) + ((lo >> 3) << 4)) * 1536 + 512 + h * 64;
    const unsigned short* kb1 = kb0 + (size_t)8 * 1536;
    union { u32x4 u[2]; bf16x16 v; } a;
    a.u[0] = *(const u32x4*)(kb0 + hi * 16);
    a.u[1] = *(const u32x4*)(kb0 + hi * 16 + 8);
    s0 = __builtin_amdgcn_wmma_f32_16x16x32_bf16(false, a.v, false, qf0, (short)0, s0, false, false);
    a.u[0] = *(const u32x4*)(kb0 + 32 + hi * 16);
    a.u[1] = *(const u32x4*)(kb0 + 40 + hi * 16);
    s0 = __builtin_amdgcn_wmma_f32_16x16x32_bf16(false, a.v, false, qf1, (short)0, s0, false, false);
    a.u[0] = *(const u32x4*)(kb1 + hi * 16);
    a.u[1] = *(const u32x4*)(kb1 + hi * 16 + 8);
    s1 = __builtin_amdgcn_wmma_f32_16x16x32_bf16(false, a.v, false, qf0, (short)0, s1, false, false);
    a.u[0] = *(const u32x4*)(kb1 + 32 + hi * 16);
    a.u[1] = *(const u32x4*)(kb1 + 40 + hi * 16);
    s1 = __builtin_amdgcn_wmma_f32_16x16x32_bf16(false, a.v, false, qf1, (short)0, s1, false, false);

    float mloc = -3.0e38f;
#pragma unroll
    for (int r = 0; r < 8; r++) {
      s0[r] *= 0.125f; s1[r] *= 0.125f;
      mloc = fmaxf(mloc, fmaxf(s0[r], s1[r]));
    }
    mloc = fmaxf(mloc, __shfl_xor(mloc, 16, 32));
    float nM = fmaxf(rM, mloc);
    float al = __expf(rM - nM); rM = nM;

    union { unsigned short s[16]; bf16x16 v; } pa;
    float ps = 0.f;
#pragma unroll
    for (int r = 0; r < 8; r++) {
      float p0 = __expf(s0[r] - nM);
      float p1 = __expf(s1[r] - nM);
      ps += p0 + p1;
      pa.s[r] = f2bf(p0);       // k_local = 16*hi + r
      pa.s[8 + r] = f2bf(p1);   // k_local = 16*hi + 8 + r
    }
    ps += __shfl_xor(ps, 16, 32);
    rS = rS * al + ps;
#pragma unroll
    for (int r = 0; r < 8; r++) {
      float ar = __shfl(al, r + 8 * hi, 32);
      o0[r] *= ar; o1[r] *= ar; o2[r] *= ar; o3[r] *= ar;
    }
    const unsigned short* vb = Vt + (size_t)((b * 8 + h) * 64 + lo) * S + kc + hi * 16;
    o0 = __builtin_amdgcn_wmma_f32_16x16x32_bf16(false, pa.v, false, *(const bf16x16*)(vb),          (short)0, o0, false, false);
    o1 = __builtin_amdgcn_wmma_f32_16x16x32_bf16(false, pa.v, false, *(const bf16x16*)(vb + 16 * S), (short)0, o1, false, false);
    o2 = __builtin_amdgcn_wmma_f32_16x16x32_bf16(false, pa.v, false, *(const bf16x16*)(vb + 32 * S), (short)0, o2, false, false);
    o3 = __builtin_amdgcn_wmma_f32_16x16x32_bf16(false, pa.v, false, *(const bf16x16*)(vb + 48 * S), (short)0, o3, false, false);
  }
  float inv = 1.f / rS;
#pragma unroll
  for (int r = 0; r < 8; r++) {
    float ir = __shfl(inv, r + 8 * hi, 32);
    size_t base = (size_t)(b * S + qt * 16 + r + 8 * hi) * 512 + h * 64 + lo;
    O[base]      = f2bf(o0[r] * ir);
    O[base + 16] = f2bf(o1[r] * ir);
    O[base + 32] = f2bf(o2[r] * ir);
    O[base + 48] = f2bf(o3[r] * ir);
  }
}

// ---------------------------------------------------------------------------
// host orchestration
// ---------------------------------------------------------------------------
struct BNP  { const float *b, *g, *m, *v; };
struct ResP { const float *b1, *b2; BNP bn1, bn2; const float *w1, *w2; };
struct EncP { BNP bn0, bn1; const float *c0b, *c0w, *c1b, *c1w, *c2b, *c2w; ResP rb0, rb1; };
struct DecP { BNP bn0, bn1; const float *d0b, *d0w, *d1b, *d1w, *d2b, *d2w; ResP rb0, rb1; };

extern "C" void kernel_launch(void* const* d_in, const int* in_sizes, int n_in,
                              void* d_out, int out_size, void* d_ws, size_t ws_size,
                              hipStream_t stream) {
  (void)in_sizes; (void)n_in; (void)out_size; (void)ws_size;
  const int Bn = 16, S = 1024, Ntok = 16384;

  // ---- unpack params in JAX pytree (sorted-key, depth-first) order --------
  int pi = 2;
  auto F    = [&]() { return (const float*)d_in[pi++]; };
  auto RBN  = [&](BNP& bn)  { bn.b = F(); bn.g = F(); bn.m = F(); bn.v = F(); };
  auto RRES = [&](ResP& r)  { r.b1 = F(); r.b2 = F(); RBN(r.bn1); RBN(r.bn2); r.w1 = F(); r.w2 = F(); };
  auto RENC = [&](EncP& e)  { RBN(e.bn0); RBN(e.bn1); e.c0b = F(); e.c0w = F(); e.c1b = F(); e.c1w = F(); e.c2b = F(); e.c2w = F(); RRES(e.rb0); RRES(e.rb1); };
  auto RDEC = [&](DecP& d)  { RBN(d.bn0); RBN(d.bn1); d.d0b = F(); d.d0w = F(); d.d1b = F(); d.d1w = F(); d.d2b = F(); d.d2w = F(); RRES(d.rb0); RRES(d.rb1); };

  const float *attn_bi = F(), *attn_bo = F(), *attn_wi = F(), *attn_wo = F();
  DecP dec_c; RDEC(dec_c);
  DecP dec_e; RDEC(dec_e);
  const float* emb_c = F();
  const float* emb_e = F();
  EncP enc_c; RENC(enc_c);
  EncP enc_e; RENC(enc_e);
  const float *post_c_b = F(), *post_c_w = F(), *post_e_b = F(), *post_e_w = F();
  const float *pre_c_b  = F(), *pre_c_w  = F(), *pre_e_b  = F(), *pre_e_w  = F();
  const float* eeg = (const float*)d_in[0];
  const float* ecg = (const float*)d_in[1];
  float* out = (float*)d_out;

  // ---- workspace bump allocator (needs ~0.9 GB) ---------------------------
  size_t off = 0;
  auto AL = [&](size_t bytes) -> char* {
    off = (off + 255) & ~(size_t)255;
    char* p = (char*)d_ws + off; off += bytes; return p;
  };
  const size_t NMAX = (size_t)512 * 32768;                   // max CN activation
  float*          P0    = (float*)AL(NMAX * 4);
  float*          P1    = (float*)AL(NMAX * 4);
  float*          DIST  = (float*)AL((size_t)1024 * Ntok * 4);
  unsigned short* IC    = (unsigned short*)AL((size_t)1536 * 65536 * 2);
  unsigned short* WSCR  = (unsigned short*)AL((size_t)1536 * 512 * 2);
  float*          SSs   = (float*)AL(1536 * 4);
  float*          SSh   = (float*)AL(1536 * 4);
  float*          ENORM = (float*)AL(1024 * 4);
  unsigned short* TMPB  = (unsigned short*)AL((size_t)Ntok * 512 * 2);
  struct SB {
    float* zf; unsigned short* zb; unsigned short* qb;
    float* postf; unsigned short* postb; unsigned short* qkv;
    unsigned short* vt; unsigned short* ob; float* decin; int* idx;
  } sb[2];
  for (int s = 0; s < 2; s++) {
    sb[s].zf    = (float*)AL((size_t)Ntok * 512 * 4);
    sb[s].zb    = (unsigned short*)AL((size_t)Ntok * 512 * 2);
    sb[s].qb    = (unsigned short*)AL((size_t)Ntok * 512 * 2);
    sb[s].postf = (float*)AL((size_t)Ntok * 512 * 4);
    sb[s].postb = (unsigned short*)AL((size_t)Ntok * 512 * 2);
    sb[s].qkv   = (unsigned short*)AL((size_t)Ntok * 1536 * 2);
    sb[s].vt    = (unsigned short*)AL((size_t)16 * 8 * 64 * 1024 * 2);
    sb[s].ob    = (unsigned short*)AL((size_t)Ntok * 512 * 2);
    sb[s].decin = (float*)AL((size_t)Ntok * 512 * 4);
    sb[s].idx   = (int*)AL((size_t)Ntok * 4);
  }

  // ---- launch helpers -----------------------------------------------------
  auto CVT = [&](const float* src, unsigned short* dst, long long n) {
    k_cvt<<<(unsigned)((n + 255) / 256), 256, 0, stream>>>(src, dst, n);
  };
  auto WTK = [&](const float* w, int I, int O, int Kk) {
    int n = I * O * Kk;
    k_wT<<<(n + 255) / 256, 256, 0, stream>>>(w, WSCR, I, O, Kk);
  };
  auto FOLD = [&](const BNP& bn, const float* cb, int C) {
    k_fold<<<(C + 63) / 64, 64, 0, stream>>>(bn.g, bn.b, bn.m, bn.v, cb, SSs, SSh, C);
  };
  auto IM2C = [&](const float* x, int C, int Kk, int st, int pad, int Lin, int Lout) {
    long long n = (long long)Bn * Lout * C * Kk;
    k_im2col<<<(unsigned)((n + 255) / 256), 256, 0, stream>>>(x, IC, C, Kk, st, pad, Lin, Lout, Bn);
  };
  auto IM2D = [&](const float* x, int C, int Kk, int padL, int Lin, int Lout) {
    long long n = (long long)Bn * Lout * C * Kk;
    k_im2col_dil<<<(unsigned)((n + 255) / 256), 256, 0, stream>>>(x, IC, C, Kk, padL, Lin, Lout, Bn);
  };
  auto GEMM = [&](const unsigned short* A, const unsigned short* Bt, int M, int N, int K,
                  const float* scale, const float* shift,
                  const float* res, int res_nc, float* outf, int outf_nc,
                  unsigned short* outb, int outb_nc, int outb_ld, int relu) {
    dim3 g(N / 256, M / 64);
    k_gemm<<<g, 256, 0, stream>>>(A, Bt, M, N, K, scale, shift, res, res_nc,
                                  outf, outf_nc, outb, outb_nc, outb_ld, relu);
  };
  auto RESB = [&](const ResP& R, int C, int Lc, float* X, float* T) {
    int Nn = Bn * Lc, Kt = C * 3;
    IM2C(X, C, 3, 1, 1, Lc, Lc);
    CVT(R.w1, WSCR, (long long)C * Kt);
    FOLD(R.bn1, R.b1, C);
    GEMM(WSCR, IC, C, Nn, Kt, SSs, SSh, nullptr, 0, T, 0, nullptr, 0, 0, 1);
    IM2C(T, C, 3, 1, 1, Lc, Lc);
    CVT(R.w2, WSCR, (long long)C * Kt);
    FOLD(R.bn2, R.b2, C);
    GEMM(WSCR, IC, C, Nn, Kt, SSs, SSh, X, 0, X, 0, nullptr, 0, 0, 1);
  };

  // zero the two loss scalars
  k_zero<<<1, 2, 0, stream>>>(out + 2 * Bn * 8192, 2);

  // ---- per-stream encoder + VQ + post ------------------------------------
  for (int s = 0; s < 2; s++) {
    const EncP&  E   = s ? enc_c : enc_e;
    const float* x   = s ? ecg : eeg;
    const float* emb = s ? emb_c : emb_e;
    const float* prw = s ? pre_c_w : pre_e_w;
    const float* prb = s ? pre_c_b : pre_e_b;
    const float* pow_ = s ? post_c_w : post_e_w;
    const float* pob  = s ? post_c_b : post_e_b;

    // conv0 + bn + lrelu
    FOLD(E.bn0, E.c0b, 256);
    {
      long long n = (long long)256 * Bn * 4096;
      k_conv0<<<(unsigned)((n + 255) / 256), 256, 0, stream>>>(x, E.c0w, SSs, SSh, P0, Bn);
    }
    RESB(E.rb0, 256, 4096, P0, P1);
    // conv1 s2
    IM2C(P0, 256, 3, 2, 1, 4096, 2048);
    CVT(E.c1w, WSCR, (long long)512 * 768);
    FOLD(E.bn1, E.c1b, 512);
    GEMM(WSCR, IC, 512, Bn * 2048, 768, SSs, SSh, nullptr, 0, P1, 0, nullptr, 0, 0, 1);
    RESB(E.rb1, 512, 2048, P1, P0);
    // conv2 s2 (bias only) -> bf16 tokens
    IM2C(P1, 512, 3, 2, 1, 2048, 1024);
    CVT(E.c2w, WSCR, (long long)512 * 1536);
    GEMM(WSCR, IC, 512, Ntok, 1536, nullptr, E.c2b, nullptr, 0, nullptr, 0, TMPB, 1, 512, 0);
    // pre 1x1 -> z (f32 + bf16, token-major)
    CVT(prw, WSCR, (long long)512 * 512);
    GEMM(WSCR, TMPB, 512, Ntok, 512, nullptr, prb, nullptr, 0, sb[s].zf, 1, sb[s].zb, 1, 512, 0);
    // VQ
    CVT(emb, WSCR, (long long)1024 * 512);
    GEMM(WSCR, sb[s].zb, 1024, Ntok, 512, nullptr, nullptr, nullptr, 0, DIST, 0, nullptr, 0, 0, 0);
    k_enorm<<<4, 256, 0, stream>>>(emb, ENORM);
    k_argmin<<<(Ntok + 255) / 256, 256, 0, stream>>>(DIST, ENORM, sb[s].idx, Ntok);
    k_gather<<<(Ntok * 512 + 255) / 256, 256, 0, stream>>>(
        sb[s].idx, emb, sb[s].zf, sb[s].qb, out + 2 * Bn * 8192 + s,
        0.25f / 8388608.0f, Ntok * 512);
    // post 1x1 -> f32 + bf16 tokens
    CVT(pow_, WSCR, (long long)512 * 512);
    GEMM(WSCR, sb[s].qb, 512, Ntok, 512, nullptr, pob, nullptr, 0,
         sb[s].postf, 1, sb[s].postb, 1, 512, 0);
    // packed QKV projection -> bf16 tokens [n][1536]
    CVT(attn_wi, WSCR, (long long)1536 * 512);
    GEMM(WSCR, sb[s].postb, 1536, Ntok, 512, nullptr, attn_bi, nullptr, 0,
         nullptr, 0, sb[s].qkv, 1, 1536, 0);
    {
      long long n = (long long)16 * 8 * 64 * 1024;
      k_vt<<<(unsigned)((n + 255) / 256), 256, 0, stream>>>(sb[s].qkv, sb[s].vt);
    }
  }

  // ---- cross attention (shared weights) -----------------------------------
  k_attn<<<1024, 256, 0, stream>>>(sb[0].qkv, sb[1].qkv, sb[1].vt, sb[0].ob);
  k_attn<<<1024, 256, 0, stream>>>(sb[1].qkv, sb[0].qkv, sb[0].vt, sb[1].ob);
  CVT(attn_wo, WSCR, (long long)512 * 512);
  GEMM(WSCR, sb[0].ob, 512, Ntok, 512, nullptr, attn_bo, sb[0].postf, 1,
       sb[0].decin, 0, nullptr, 0, 0, 0);
  GEMM(WSCR, sb[1].ob, 512, Ntok, 512, nullptr, attn_bo, sb[1].postf, 1,
       sb[1].decin, 0, nullptr, 0, 0, 0);

  // ---- decoders -----------------------------------------------------------
  for (int s = 0; s < 2; s++) {
    const DecP& D = s ? dec_c : dec_e;
    float* seg = out + (size_t)s * Bn * 8192;
    // d0: ConvT 512->512 K3 s2 (L 1024 -> 2048)
    IM2D(sb[s].decin, 512, 3, 1, 1024, 2048);
    WTK(D.d0w, 512, 512, 3);
    FOLD(D.bn0, D.d0b, 512);
    GEMM(WSCR, IC, 512, Bn * 2048, 1536, SSs, SSh, nullptr, 0, P0, 0, nullptr, 0, 0, 1);
    RESB(D.rb0, 512, 2048, P0, P1);
    // d1: ConvT 512->256 K3 s2 (L 2048 -> 4096)
    IM2D(P0, 512, 3, 1, 2048, 4096);
    WTK(D.d1w, 512, 256, 3);
    FOLD(D.bn1, D.d1b, 256);
    GEMM(WSCR, IC, 256, Bn * 4096, 1536, SSs, SSh, nullptr, 0, P1, 0, nullptr, 0, 0, 1);
    RESB(D.rb1, 256, 4096, P1, P0);
    // d2: ConvT 256->1 K5 s2 (L 4096 -> 8192), direct
    k_d2<<<(Bn * 8192 + 255) / 256, 256, 0, stream>>>(P1, D.d2w, D.d2b, seg, Bn);
  }
}
